// AHBF_non_43963285242135
// MI455X (gfx1250) — compile-verified, matrix operand order checked by reading, not run on hardware
//
#include <hip/hip_runtime.h>
#include <hip/hip_bf16.h>
#include <math.h>

// Problem constants (from reference): B=8, C=64, H=W=64, inter=32, N=H*W/2=2048
#define BB    8
#define CC    64
#define HW    4096
#define NN    2048
#define NCLS  100
#define EPSV  1e-5f

typedef __attribute__((ext_vector_type(16))) __bf16 v16bf;
typedef __attribute__((ext_vector_type(8)))  __bf16 v8bf;
typedef __attribute__((ext_vector_type(8)))  float  v8f;

// ---------------------------------------------------------------------------
// WMMA fragment helpers (CDNA5 wave32 layouts, CDNA5 ISA §7.12.2)
// A (16x32 bf16): lane L<16 holds row M=L, K=0..7 & 16..23; lane L+16 holds the
// same row, K=8..15 & 24..31.  B (32x16): symmetric with lane = column, so when
// B is stored transposed (column-major => row-major of B^T) the per-lane access
// pattern is identical to A.  Both halves are contiguous 16-byte chunks =>
// two global_load_b128 per fragment.
// C/D: 8 VGPRs; lanes 0-15 -> N=lane, M=v; lanes 16-31 -> N=lane-16, M=v+8.
// ---------------------------------------------------------------------------
__device__ inline v16bf frag_contig(const __bf16* __restrict__ base,
                                    int row0, int ld, int k0, int lane) {
  int r  = lane & 15;
  int kh = (lane >> 4) << 3;             // 0 or 8
  const __bf16* p = base + (size_t)(row0 + r) * ld + k0 + kh;
  union { v16bf v; struct { v8bf lo, hi; } h; } u;
  u.h.lo = *(const v8bf*)(p);            // K = kh..kh+7
  u.h.hi = *(const v8bf*)(p + 16);       // K = 16+kh..23+kh
  return u.v;
}

__device__ inline void store_cfrag(float* __restrict__ out, int row0, int col0,
                                   int ldo, int lane, v8f c) {
  int col   = col0 + (lane & 15);
  int rbase = row0 + ((lane >> 4) << 3);
#pragma unroll
  for (int v = 0; v < 8; ++v) out[(size_t)(rbase + v) * ldo + col] = c[v];
}

// ---------------------------------------------------------------------------
// Stage 1: conv1 (1x1, 128->64) + per-channel sum/sumsq for BN
// ---------------------------------------------------------------------------
__global__ void k_init_stats(float* stats) {
  int t = threadIdx.x;
  if (t < 128) stats[t] = 0.f;
}

__global__ void k_conv1(const float* __restrict__ x, const float* __restrict__ y,
                        const float* __restrict__ w, const float* __restrict__ bias,
                        float* __restrict__ z, float* __restrict__ stats) {
  __shared__ float wrow[128];
  __shared__ float red[256];
  int bo = blockIdx.x;               // b*64 + o
  int b = bo >> 6, o = bo & 63;
  int tid = threadIdx.x;
  if (tid < 128) wrow[tid] = w[o * 128 + tid];
  __syncthreads();
  const float* xb = x + (size_t)b * CC * HW;
  const float* yb = y + (size_t)b * CC * HW;
  float bi = bias[o];
  float bsum = 0.f, bsq = 0.f;
  for (int s = tid; s < HW; s += 256) {
    float acc = bi;
#pragma unroll 8
    for (int c = 0; c < 64; ++c) acc += wrow[c] * xb[(size_t)c * HW + s];
#pragma unroll 8
    for (int c = 0; c < 64; ++c) acc += wrow[64 + c] * yb[(size_t)c * HW + s];
    z[(size_t)bo * HW + s] = acc;
    bsum += acc; bsq += acc * acc;
  }
  red[tid] = bsum; __syncthreads();
  for (int st = 128; st > 0; st >>= 1) { if (tid < st) red[tid] += red[tid + st]; __syncthreads(); }
  float tot = red[0]; __syncthreads();
  red[tid] = bsq; __syncthreads();
  for (int st = 128; st > 0; st >>= 1) { if (tid < st) red[tid] += red[tid + st]; __syncthreads(); }
  if (tid == 0) { atomicAdd(&stats[o], tot); atomicAdd(&stats[64 + o], red[0]); }
}

__global__ void k_bn_finalize(float* __restrict__ stats, const float* __restrict__ g,
                              const float* __restrict__ beta) {
  int o = threadIdx.x;
  if (o < 64) {
    float inv_m = 1.f / (float)(BB * HW);
    float mean = stats[o] * inv_m;
    float var  = stats[64 + o] * inv_m - mean * mean;
    float sc   = g[o] * rsqrtf(var + EPSV);
    stats[128 + o] = sc;
    stats[192 + o] = beta[o] - mean * sc;
  }
}

__global__ void k_bn_apply(float* __restrict__ z, const float* __restrict__ stats) {
  size_t i = (size_t)blockIdx.x * 256 + threadIdx.x;
  int o = (int)((i >> 12) & 63);               // layout (b*64+o)*4096+s
  z[i] = z[i] * stats[128 + o] + stats[192 + o];
}

// ---------------------------------------------------------------------------
// Stage 2: phi/theta/g projections (32x64 each) in bf16 WMMA-friendly layouts.
//   viewed channel r = 2*i + (s>=2048), column n = s & 2047
//   Ptr, Ttr : [b][n][r]  (2048 x 64)  -- transposed view (contiguous-K frags)
//   Gm       : [b][r][n]  (64 x 2048)  -- viewed row-major (B^T frags, K = n)
// ---------------------------------------------------------------------------
__global__ void k_proj(const float* __restrict__ z,
                       const float* __restrict__ phi_w, const float* __restrict__ theta_w,
                       const float* __restrict__ g_w,
                       __bf16* __restrict__ Ptr, __bf16* __restrict__ Ttr,
                       __bf16* __restrict__ Gm) {
  __shared__ float Wsh[32 * 64];
  int proj = blockIdx.z;
  const float* Wp = (proj == 0) ? phi_w : (proj == 1) ? theta_w : g_w;
  for (int i = threadIdx.x; i < 32 * 64; i += 256) Wsh[i] = Wp[i];
  __syncthreads();
  int b = blockIdx.y;
  int s = blockIdx.x * 256 + threadIdx.x;
  const float* f = z + (size_t)b * CC * HW + s;
  float acc[32];
#pragma unroll
  for (int i = 0; i < 32; ++i) acc[i] = 0.f;
  for (int c = 0; c < 64; ++c) {
    float v = f[(size_t)c * HW];
#pragma unroll
    for (int i = 0; i < 32; ++i) acc[i] += Wsh[i * 64 + c] * v;
  }
  int n = s & (NN - 1), shi = s >> 11;
  if (proj == 2) {
    __bf16* dst = Gm + (size_t)b * CC * NN;
#pragma unroll
    for (int i = 0; i < 32; ++i) dst[(size_t)(2 * i + shi) * NN + n] = (__bf16)acc[i];
  } else {
    __bf16* dst = ((proj == 0) ? Ptr : Ttr) + (size_t)b * NN * CC;
#pragma unroll
    for (int i = 0; i < 32; ++i) dst[(size_t)n * CC + (2 * i + shi)] = (__bf16)acc[i];
  }
}

// ---------------------------------------------------------------------------
// Stage 3a: attn[n][m] = sum_c Ttr[n][c] * Ptr[m][c]   (M=N=2048, K=64)
// 128 threads = 4 waves; block tile 64x64; wave -> 16 rows x 64 cols.
// All fragments are contiguous 2x b128 loads per lane.
// ---------------------------------------------------------------------------
__global__ void k_gemm_attn(const __bf16* __restrict__ T, const __bf16* __restrict__ P,
                            float* __restrict__ attn) {
  int lane = threadIdx.x & 31, wv = threadIdx.x >> 5;
  int n0 = blockIdx.y * 64 + wv * 16;
  int m0 = blockIdx.x * 64;
  v8f c0 = {}, c1 = {}, c2 = {}, c3 = {};
#pragma unroll
  for (int kb = 0; kb < 64; kb += 32) {
    v16bf a  = frag_contig(T, n0,      CC, kb, lane);
    v16bf b0 = frag_contig(P, m0,      CC, kb, lane);
    v16bf b1 = frag_contig(P, m0 + 16, CC, kb, lane);
    v16bf b2 = frag_contig(P, m0 + 32, CC, kb, lane);
    v16bf b3 = frag_contig(P, m0 + 48, CC, kb, lane);
    c0 = __builtin_amdgcn_wmma_f32_16x16x32_bf16(false, a, false, b0, (short)0, c0, false, false);
    c1 = __builtin_amdgcn_wmma_f32_16x16x32_bf16(false, a, false, b1, (short)0, c1, false, false);
    c2 = __builtin_amdgcn_wmma_f32_16x16x32_bf16(false, a, false, b2, (short)0, c2, false, false);
    c3 = __builtin_amdgcn_wmma_f32_16x16x32_bf16(false, a, false, b3, (short)0, c3, false, false);
  }
  store_cfrag(attn, n0, m0,      NN, lane, c0);
  store_cfrag(attn, n0, m0 + 16, NN, lane, c1);
  store_cfrag(attn, n0, m0 + 32, NN, lane, c2);
  store_cfrag(attn, n0, m0 + 48, NN, lane, c3);
}

// Stage 3b: softmax over axis n (per column m): column max + sum of exp
__global__ void k_colstats(const float* __restrict__ attn,
                           float* __restrict__ colmax, float* __restrict__ colsum) {
  __shared__ float red[256];
  int m = blockIdx.x, tid = threadIdx.x;
  float mx = -1e30f;
  for (int n = tid; n < NN; n += 256) mx = fmaxf(mx, attn[(size_t)n * NN + m]);
  red[tid] = mx; __syncthreads();
  for (int st = 128; st > 0; st >>= 1) { if (tid < st) red[tid] = fmaxf(red[tid], red[tid + st]); __syncthreads(); }
  float cm = red[0]; __syncthreads();
  float sm = 0.f;
  for (int n = tid; n < NN; n += 256) sm += expf(attn[(size_t)n * NN + m] - cm);
  red[tid] = sm; __syncthreads();
  for (int st = 128; st > 0; st >>= 1) { if (tid < st) red[tid] += red[tid + st]; __syncthreads(); }
  if (tid == 0) { colmax[m] = cm; colsum[m] = red[0]; }
}

// Stage 3c: normalized probabilities -> bf16
__global__ void k_softnorm(const float* __restrict__ attn, const float* __restrict__ colmax,
                           const float* __restrict__ colsum, __bf16* __restrict__ S) {
  size_t i = (size_t)blockIdx.x * 256 + threadIdx.x;
  int m = (int)(i & (NN - 1));
  S[i] = (__bf16)(expf(attn[i] - colmax[m]) / colsum[m]);
}

// Stage 3d: o[n][c] = sum_m S[n][m] * Gm[c][m]   (M=2048, N=64, K=2048)
__global__ void k_gemm_o(const __bf16* __restrict__ S, const __bf16* __restrict__ G,
                         float* __restrict__ o) {
  int lane = threadIdx.x & 31, wv = threadIdx.x >> 5;
  int n0 = blockIdx.x * 64 + wv * 16;
  v8f c0 = {}, c1 = {}, c2 = {}, c3 = {};
  for (int kb = 0; kb < NN; kb += 32) {
    if (kb + 64 <= NN - 32) {   // uniform: prefetch next K panel of S
      __builtin_prefetch(S + (size_t)(n0 + (lane & 15)) * NN + kb + 64, 0, 0);
    }
    v16bf a  = frag_contig(S, n0, NN, kb, lane);
    v16bf b0 = frag_contig(G, 0,  NN, kb, lane);
    v16bf b1 = frag_contig(G, 16, NN, kb, lane);
    v16bf b2 = frag_contig(G, 32, NN, kb, lane);
    v16bf b3 = frag_contig(G, 48, NN, kb, lane);
    c0 = __builtin_amdgcn_wmma_f32_16x16x32_bf16(false, a, false, b0, (short)0, c0, false, false);
    c1 = __builtin_amdgcn_wmma_f32_16x16x32_bf16(false, a, false, b1, (short)0, c1, false, false);
    c2 = __builtin_amdgcn_wmma_f32_16x16x32_bf16(false, a, false, b2, (short)0, c2, false, false);
    c3 = __builtin_amdgcn_wmma_f32_16x16x32_bf16(false, a, false, b3, (short)0, c3, false, false);
  }
  store_cfrag(o, n0, 0,  CC, lane, c0);
  store_cfrag(o, n0, 16, CC, lane, c1);
  store_cfrag(o, n0, 32, CC, lane, c2);
  store_cfrag(o, n0, 48, CC, lane, c3);
}

// ---------------------------------------------------------------------------
// Stage 4: mask conv (32->64) + residual add + spatial mean pool
//   o32[ci][s] = o[n = s&2047][c = 2*ci + (s>>11)]
// ---------------------------------------------------------------------------
__global__ void k_mask_pool(const float* __restrict__ feasc, const float* __restrict__ obuf,
                            const float* __restrict__ mask_w, float* __restrict__ out,
                            float* __restrict__ pooled) {
  __shared__ float mw[32];
  __shared__ float red[256];
  int bo = blockIdx.x, tid = threadIdx.x;
  int b = bo >> 6, oc = bo & 63;
  if (tid < 32) mw[tid] = mask_w[oc * 32 + tid];
  __syncthreads();
  const float* ob = obuf + (size_t)b * NN * CC;
  float ls = 0.f;
  for (int s = tid; s < HW; s += 256) {
    int n = s & (NN - 1), shi = s >> 11;
    const float* orow = ob + (size_t)n * CC + shi;
    float m = 0.f;
#pragma unroll
    for (int ci = 0; ci < 32; ++ci) m += mw[ci] * orow[2 * ci];
    float v = feasc[(size_t)bo * HW + s] + m;
    out[(size_t)bo * HW + s] = v;
    ls += v;
  }
  red[tid] = ls; __syncthreads();
  for (int st = 128; st > 0; st >>= 1) { if (tid < st) red[tid] += red[tid + st]; __syncthreads(); }
  if (tid == 0) pooled[bo] = red[0] * (1.f / (float)HW);
}

// Stage 5: gating head (fc -> BN1d -> relu -> softmax) + logit blend
__global__ void k_gate_logit(const float* __restrict__ pooled,
                             const float* __restrict__ fc_w, const float* __restrict__ fc_b,
                             const float* __restrict__ bnv_g, const float* __restrict__ bnv_b,
                             const float* __restrict__ logitx, const float* __restrict__ logity,
                             float* __restrict__ logit_out) {
  __shared__ float feas[BB * 2];
  __shared__ float wgt[BB * 2];
  int tid = threadIdx.x;
  if (tid < BB * 2) {
    int b = tid >> 1, j = tid & 1;
    float a = fc_b[j];
    for (int c = 0; c < 64; ++c) a += pooled[b * 64 + c] * fc_w[j * 64 + c];
    feas[tid] = a;
  }
  __syncthreads();
  if (tid < 2) {
    int j = tid;
    float mn = 0.f;
    for (int b = 0; b < BB; ++b) mn += feas[b * 2 + j];
    mn *= (1.f / BB);
    float vr = 0.f;
    for (int b = 0; b < BB; ++b) { float d = feas[b * 2 + j] - mn; vr += d * d; }
    vr *= (1.f / BB);
    float sc = bnv_g[j] * rsqrtf(vr + EPSV);
    for (int b = 0; b < BB; ++b)
      feas[b * 2 + j] = fmaxf(0.f, (feas[b * 2 + j] - mn) * sc + bnv_b[j]);
  }
  __syncthreads();
  if (tid < BB) {
    float r0 = feas[tid * 2], r1 = feas[tid * 2 + 1];
    float mx = fmaxf(r0, r1);
    float e0 = expf(r0 - mx), e1 = expf(r1 - mx);
    float inv = 1.f / (e0 + e1);
    wgt[tid * 2] = e0 * inv; wgt[tid * 2 + 1] = e1 * inv;
  }
  __syncthreads();
  for (int i = tid; i < BB * NCLS; i += 256) {
    int b = i / NCLS;
    logit_out[i] = wgt[b * 2] * logitx[i] + wgt[b * 2 + 1] * logity[i];
  }
}

// ---------------------------------------------------------------------------
extern "C" void kernel_launch(void* const* d_in, const int* in_sizes, int n_in,
                              void* d_out, int out_size, void* d_ws, size_t ws_size,
                              hipStream_t stream) {
  (void)in_sizes; (void)n_in; (void)out_size; (void)ws_size;
  const float* x       = (const float*)d_in[0];
  const float* y       = (const float*)d_in[1];
  const float* logitx  = (const float*)d_in[2];
  const float* logity  = (const float*)d_in[3];
  const float* conv1_w = (const float*)d_in[4];
  const float* conv1_b = (const float*)d_in[5];
  const float* bn1_g   = (const float*)d_in[6];
  const float* bn1_b   = (const float*)d_in[7];
  const float* phi_w   = (const float*)d_in[8];
  const float* theta_w = (const float*)d_in[9];
  const float* g_w     = (const float*)d_in[10];
  const float* mask_w  = (const float*)d_in[11];
  const float* fc_w    = (const float*)d_in[12];
  const float* fc_b    = (const float*)d_in[13];
  const float* bnv_g   = (const float*)d_in[14];
  const float* bnv_b   = (const float*)d_in[15];

  // workspace layout (~44 MB; attention buffers reused per batch)
  float*  z      = (float*)d_ws;                         // B*C*HW
  float*  stats  = z + (size_t)BB * CC * HW;             // 256
  __bf16* Ttr    = (__bf16*)(stats + 256);               // B*N*64
  __bf16* Ptr    = Ttr + (size_t)BB * NN * CC;           // B*N*64
  __bf16* Gm     = Ptr + (size_t)BB * NN * CC;           // B*64*N
  float*  attn   = (float*)(Gm + (size_t)BB * CC * NN);  // N*N (per-batch)
  float*  colmax = attn + (size_t)NN * NN;               // N
  float*  colsum = colmax + NN;                          // N
  __bf16* Sbf    = (__bf16*)(colsum + NN);               // N*N (per-batch)
  float*  obuf   = (float*)(Sbf + (size_t)NN * NN);      // B*N*64
  float*  pooled = obuf + (size_t)BB * NN * CC;          // B*64

  float* out_feasc = (float*)d_out;
  float* out_logit = out_feasc + (size_t)BB * CC * HW;

  k_init_stats<<<1, 128, 0, stream>>>(stats);
  k_conv1<<<BB * CC, 256, 0, stream>>>(x, y, conv1_w, conv1_b, z, stats);
  k_bn_finalize<<<1, 64, 0, stream>>>(stats, bn1_g, bn1_b);
  k_bn_apply<<<(BB * CC * HW) / 256, 256, 0, stream>>>(z, stats);
  k_proj<<<dim3(HW / 256, BB, 3), 256, 0, stream>>>(z, phi_w, theta_w, g_w, Ptr, Ttr, Gm);

  for (int b = 0; b < BB; ++b) {
    k_gemm_attn<<<dim3(NN / 64, NN / 64), 128, 0, stream>>>(
        Ttr + (size_t)b * NN * CC, Ptr + (size_t)b * NN * CC, attn);
    k_colstats<<<NN, 256, 0, stream>>>(attn, colmax, colsum);
    k_softnorm<<<(NN * NN) / 256, 256, 0, stream>>>(attn, colmax, colsum, Sbf);
    k_gemm_o<<<NN / 64, 128, 0, stream>>>(Sbf, Gm + (size_t)b * CC * NN,
                                          obuf + (size_t)b * NN * CC);
  }

  k_mask_pool<<<BB * CC, 256, 0, stream>>>(z, obuf, mask_w, out_feasc, pooled);
  k_gate_logit<<<1, 256, 0, stream>>>(pooled, fc_w, fc_b, bnv_g, bnv_b,
                                      logitx, logity, out_logit);
}
